// Attention_layer_32762010534086
// MI455X (gfx1250) — matrix operand order repaired and checked by reading, not compiled, not run on hardware
//
#include <hip/hip_runtime.h>

typedef __bf16 bf16_t;
typedef __bf16 v16bf __attribute__((ext_vector_type(16)));
typedef __bf16 v8bf  __attribute__((ext_vector_type(8)));
typedef float  v8f   __attribute__((ext_vector_type(8)));

// Problem dims (fixed by reference)
#define BB 8
#define CC 128
#define DD 64
#define HH 32
#define WW 32
#define SS 64
#define DHW  (DD * HH * WW)       // 65536
#define CDHW (CC * DHW)           // 8388608
#define NDHW (BB * DHW)           // 524288 elements per channel for BN stats

// ---- B-operand panel layout --------------------------------------------------
// For a [K][N] matrix consumed as WMMA B: split into (K/32)x(N/16) panels.
// Within a panel, lane l = (k&16) + (n&15) owns 16 contiguous bf16 (k&15 = 0..15).
// Lane stride padded to 24 elems (48 B) so each 16B half-run is 16B-aligned and
// lanes spread across banks. Panel stride = 32*24 = 768 elems.
#define BLSTR 24
#define BPANEL (32 * BLSTR)       // 768 elems per panel

// Row-major strides for A-operand sources (elements; rows are 16B multiples)
#define WLDP 136  // W proj staging [64][WLDP]   (272 B rows)
#define WLDO 72   // Wo staging     [128][WLDO]  (144 B rows)
#define KLD  72   // Kt / V row-major [64][KLD]  (144 B rows)
#define SLD  68   // scores f32 [64][SLD]

// LDS byte offsets
#define OFF_X  0                  // X panels: (128/32)*(64/16)=16 panels -> 24576 B
#define OFF_W  24576              // staging: max(64*136, 128*72)=9216 elems -> 18432 B
#define OFF_KT 43008              // Kt row-major [64][72] -> 9216 B
#define OFF_Q  52224              // Q panels: 8 panels -> 12288 B  (reused by attn panels)
#define OFF_V  64512              // V row-major [64][72] -> 9216 B
#define OFF_SC 73728              // scores f32 [64][68]  -> 17408 B
#define OFF_AB 91136              // softmax panels: 8 panels -> 12288 B
#define LDS_BYTES 103424

// ---------------- fragment helpers (CDNA5 wave32 16-bit layouts) ----------------

// A-matrix 16x32 (MxK) from row-major source, row stride ld (16B-multiple rows).
// Per lane: two contiguous 8-element runs -> compiler emits 2x ds_load_b128.
__device__ __forceinline__ v16bf load_a16x32(const bf16_t* base, int ld, int lane) {
  const int m  = lane & 15;
  const int k8 = (lane & 16) ? 16 : 0;
  const bf16_t* row = base + m * ld + k8;
  v8bf lo = *(const v8bf*)(row);        // K = k8 + 0..7
  v8bf hi = *(const v8bf*)(row + 8);    // K = k8 + 8..15
  return __builtin_shufflevector(lo, hi, 0, 1, 2, 3, 4, 5, 6, 7,
                                 8, 9, 10, 11, 12, 13, 14, 15);
}

// B-matrix 32x16 fragment from panel storage: one 32B run per lane (2x b128).
__device__ __forceinline__ v16bf load_b_frag(const bf16_t* pan, int nt16, int kt,
                                             int nt, int lane) {
  const bf16_t* p = pan + ((kt * nt16 + nt) * 32 + lane) * BLSTR;
  v8bf lo = *(const v8bf*)(p);
  v8bf hi = *(const v8bf*)(p + 8);
  return __builtin_shufflevector(lo, hi, 0, 1, 2, 3, 4, 5, 6, 7,
                                 8, 9, 10, 11, 12, 13, 14, 15);
}

// Store a f32 16x16 D-tile (tile m16,n16 of a [M][N] matrix) into B-panel layout
// as bf16: per lane one packed 16B ds_store_b128.
__device__ __forceinline__ void store_d_to_b_panel(bf16_t* pan, int nt16, int m16,
                                                   int n16, int lane, v8f acc) {
  const int d  = n16 * 16 + (lane & 15);            // N coordinate
  const int s0 = m16 * 16 + ((lane & 16) ? 8 : 0);  // first M (K-of-panel) coord
  const int lane_b = (s0 & 16) + (d & 15);
  bf16_t* p = pan + (((s0 >> 5) * nt16 + (d >> 4)) * 32 + lane_b) * BLSTR + (s0 & 15);
  v8bf pk;
#pragma unroll
  for (int r = 0; r < 8; ++r) pk[r] = (bf16_t)acc[r];
  *(v8bf*)p = pk;
}

__device__ __forceinline__ v8f wmma_bf16(v16bf a, v16bf b, v8f c) {
  return __builtin_amdgcn_wmma_f32_16x16x32_bf16(false, a, false, b, (short)0, c,
                                                 false, false);
}

// ---------------- kernel 1: BatchNorm stats -> fused scale/shift ----------------

__global__ __launch_bounds__(256) void bn_stats_kernel(
    const float* __restrict__ x, const float* __restrict__ gamma,
    const float* __restrict__ beta, float* __restrict__ sA, float* __restrict__ sB) {
  const int c   = blockIdx.x;          // one block per channel
  const int tid = threadIdx.x;
  float s = 0.f, ss = 0.f;
  for (int e = tid; e < NDHW; e += 256) {
    const int b = e >> 16;             // DHW = 65536
    const int r = e & (DHW - 1);
    const float v = x[(long long)b * CDHW + (long long)c * DHW + r];
    s += v;
    ss += v * v;
  }
  __shared__ float rs[256], rss[256];
  rs[tid] = s; rss[tid] = ss;
  __syncthreads();
  for (int o = 128; o > 0; o >>= 1) {
    if (tid < o) { rs[tid] += rs[tid + o]; rss[tid] += rss[tid + o]; }
    __syncthreads();
  }
  if (tid == 0) {
    const float inv_n = 1.0f / (float)NDHW;
    const float mean  = rs[0] * inv_n;
    const float var   = rss[0] * inv_n - mean * mean;
    const float rstd  = rsqrtf(var + 1e-5f);
    const float g     = gamma[c] * rstd;
    sA[c] = g;
    sB[c] = beta[c] - mean * g;
  }
}

// ---------------- kernel 2: weights fp32 -> bf16 (contiguous in workspace) -----

__global__ __launch_bounds__(256) void cvt_weights_kernel(
    const float* __restrict__ Wk, const float* __restrict__ Wq,
    const float* __restrict__ Wv, const float* __restrict__ Wo,
    bf16_t* __restrict__ outw) {
  const int i = blockIdx.x * 256 + threadIdx.x;   // 32768 total
  const float* src;
  int off;
  if (i < 8192)       { src = Wk; off = 0; }
  else if (i < 16384) { src = Wq; off = 8192; }
  else if (i < 24576) { src = Wv; off = 16384; }
  else                { src = Wo; off = 24576; }
  outw[i] = (bf16_t)src[i - off];
}

// ---------------- kernel 3: fused per-site attention ----------------------------

__global__ __launch_bounds__(256) void attn_fused_kernel(
    const float* __restrict__ x, const float* __restrict__ sA,
    const float* __restrict__ sB, const bf16_t* __restrict__ wkb,
    const bf16_t* __restrict__ wqb, const bf16_t* __restrict__ wvb,
    const bf16_t* __restrict__ wob, float* __restrict__ out) {
  extern __shared__ char smem[];
  bf16_t* Xp = (bf16_t*)(smem + OFF_X);    // X in B-panels (K=c 128, N=d 64)
  bf16_t* Wb = (bf16_t*)(smem + OFF_W);    // weight staging, row-major (A source)
  bf16_t* Kt = (bf16_t*)(smem + OFF_KT);   // K transposed [d][s] row-major (A source)
  bf16_t* Qp = (bf16_t*)(smem + OFF_Q);    // Q in B-panels (K=s 64, N=d 64)
  bf16_t* Vs = (bf16_t*)(smem + OFF_V);    // V [s][d] row-major (A source)
  float*  Sc = (float*)(smem + OFF_SC);    // scores [i][j] f32
  bf16_t* Ap = (bf16_t*)(smem + OFF_AB);   // softmax(A) in B-panels
  bf16_t* Tp = Qp;                         // attn result panels reuse Q storage

  const int tid  = threadIdx.x;
  const int lane = tid & 31;
  const int wv   = tid >> 5;               // 8 wave32 waves

  const int n  = blockIdx.x;               // site index = b*H*W + h*W + w
  const int b  = n >> 10;
  const int hw = n & 1023;
  const int h  = hw >> 5;
  const int w  = hw & 31;
  const long long base = (long long)b * CDHW + h * WW + w;

  // ---- Stage A: gather x tile, fused BN scale/shift, store to B panels ----
  for (int e = tid; e < CC * DD; e += 256) {
    const int c = e >> 6, d = e & 63;
    const float xv = x[base + (long long)c * DHW + d * (HH * WW)];
    const int lane_b = (c & 16) + (d & 15);
    Xp[(((c >> 5) * 4 + (d >> 4)) * 32 + lane_b) * BLSTR + (c & 15)] =
        (bf16_t)(xv * sA[c] + sB[c]);
  }

  // ---- Stage B/C: three projections W(64x128) * X(128x64) via WMMA ----
  for (int p = 0; p < 3; ++p) {
    __syncthreads();                       // protect Wb reuse across p
    const bf16_t* Wp = (p == 0) ? wkb : ((p == 1) ? wqb : wvb);
    for (int e = tid; e < 1024; e += 256)  // 16B-granular staging
      *(v8bf*)(Wb + (e >> 4) * WLDP + (e & 15) * 8) = *(const v8bf*)(Wp + e * 8);
    __syncthreads();                       // Wb (and Xp on p==0) ready
#pragma unroll
    for (int ti = 0; ti < 2; ++ti) {       // 2 output tiles per wave
      const int t = ti * 8 + wv;
      const int m16 = t >> 2, n16 = t & 3;
      v8f acc = {};
#pragma unroll
      for (int kk = 0; kk < 4; ++kk) {     // K = 128 in steps of 32
        v16bf a  = load_a16x32(Wb + m16 * 16 * WLDP + kk * 32, WLDP, lane);
        v16bf bf = load_b_frag(Xp, 4, kk, n16, lane);
        acc = wmma_bf16(a, bf, acc);
      }
      if (p == 0) {                        // K stored transposed: Kt[d][s] (A source)
        const int nn = n16 * 16 + (lane & 15);
        const int mo = m16 * 16 + ((lane & 16) ? 8 : 0);
#pragma unroll
        for (int r = 0; r < 8; ++r) Kt[nn * KLD + (mo + r)] = (bf16_t)acc[r];
      } else if (p == 1) {                 // Q -> B panels (one b128 store)
        store_d_to_b_panel(Qp, 4, m16, n16, lane, acc);
      } else {                             // V row-major [s][d] (A source)
        const int nn = n16 * 16 + (lane & 15);
        const int mo = m16 * 16 + ((lane & 16) ? 8 : 0);
#pragma unroll
        for (int r = 0; r < 8; ++r) Vs[(mo + r) * KLD + nn] = (bf16_t)acc[r];
      }
    }
  }
  __syncthreads();

  // ---- Stage D: scores[i][j] = sum_s K[s][i]*Q[s][j] * (1/sqrt(64)) ----
#pragma unroll
  for (int ti = 0; ti < 2; ++ti) {
    const int t = ti * 8 + wv;
    const int m16 = t >> 2, n16 = t & 3;
    v8f acc = {};
#pragma unroll
    for (int kk = 0; kk < 2; ++kk) {       // K = 64 in steps of 32
      v16bf a  = load_a16x32(Kt + m16 * 16 * KLD + kk * 32, KLD, lane);
      v16bf bf = load_b_frag(Qp, 4, kk, n16, lane);
      acc = wmma_bf16(a, bf, acc);
    }
    const int nn = n16 * 16 + (lane & 15);
    const int mo = m16 * 16 + ((lane & 16) ? 8 : 0);
#pragma unroll
    for (int r = 0; r < 8; ++r) Sc[(mo + r) * SLD + nn] = acc[r] * 0.125f;
  }
  __syncthreads();

  // ---- Stage E: softmax over key axis i, per column j (fp32) ----
  if (tid < 64) {
    const int j = tid;
    float mx = -3.0e38f;
    for (int i = 0; i < 64; ++i) mx = fmaxf(mx, Sc[i * SLD + j]);
    float sum = 0.f;
    for (int i = 0; i < 64; ++i) sum += __expf(Sc[i * SLD + j] - mx);
    const float inv = 1.0f / sum;
    for (int i = 0; i < 64; ++i) {
      const float av = __expf(Sc[i * SLD + j] - mx) * inv;
      const int lane_b = (i & 16) + (j & 15);
      Ap[(((i >> 5) * 4 + (j >> 4)) * 32 + lane_b) * BLSTR + (i & 15)] = (bf16_t)av;
    }
  }
  __syncthreads();

  // ---- Stage F: attn = V(64x64) * A(64x64) -> Tp panels; stage Wo into Wb ----
  for (int e = tid; e < 1024; e += 256)    // Wo: [128][64] -> row-major [128][WLDO]
    *(v8bf*)(Wb + (e >> 3) * WLDO + (e & 7) * 8) = *(const v8bf*)(wob + e * 8);
#pragma unroll
  for (int ti = 0; ti < 2; ++ti) {
    const int t = ti * 8 + wv;
    const int m16 = t >> 2, n16 = t & 3;
    v8f acc = {};
#pragma unroll
    for (int kk = 0; kk < 2; ++kk) {
      v16bf a  = load_a16x32(Vs + m16 * 16 * KLD + kk * 32, KLD, lane);
      v16bf bf = load_b_frag(Ap, 4, kk, n16, lane);
      acc = wmma_bf16(a, bf, acc);
    }
    store_d_to_b_panel(Tp, 4, m16, n16, lane, acc);   // attn -> B panels
  }
  __syncthreads();

  // ---- Stage G: out = Wo(128x64) * attn(64x64); residual add; scatter ----
#pragma unroll
  for (int ti = 0; ti < 4; ++ti) {         // 4 output tiles per wave
    const int t = ti * 8 + wv;
    const int m16 = t >> 2, n16 = t & 3;
    v8f acc = {};
#pragma unroll
    for (int kk = 0; kk < 2; ++kk) {
      v16bf a  = load_a16x32(Wb + m16 * 16 * WLDO + kk * 32, WLDO, lane);
      v16bf bf = load_b_frag(Tp, 4, kk, n16, lane);
      acc = wmma_bf16(a, bf, acc);
    }
    const int d  = n16 * 16 + (lane & 15);
    const int c0 = m16 * 16 + ((lane & 16) ? 8 : 0);
#pragma unroll
    for (int r = 0; r < 8; ++r) {
      const long long gi = base + (long long)(c0 + r) * DHW + (long long)d * (HH * WW);
      out[gi] = x[gi] + acc[r];
    }
  }
}

// ---------------- host-side launch ----------------------------------------------

extern "C" void kernel_launch(void* const* d_in, const int* in_sizes, int n_in,
                              void* d_out, int out_size, void* d_ws, size_t ws_size,
                              hipStream_t stream) {
  (void)in_sizes; (void)n_in; (void)out_size; (void)ws_size;
  const float* x     = (const float*)d_in[0];
  const float* Wk    = (const float*)d_in[1];
  const float* Wq    = (const float*)d_in[2];
  const float* Wv    = (const float*)d_in[3];
  const float* Wo    = (const float*)d_in[4];
  const float* gamma = (const float*)d_in[5];
  const float* beta  = (const float*)d_in[6];
  float* out = (float*)d_out;

  char* ws = (char*)d_ws;
  float*  sA  = (float*)ws;            // 128 f32
  float*  sB  = (float*)(ws + 512);    // 128 f32
  bf16_t* wkb = (bf16_t*)(ws + 1024);  // 4 x 8192 bf16
  bf16_t* wqb = wkb + 8192;
  bf16_t* wvb = wqb + 8192;
  bf16_t* wob = wvb + 8192;

  bn_stats_kernel<<<CC, 256, 0, stream>>>(x, gamma, beta, sA, sB);
  cvt_weights_kernel<<<128, 256, 0, stream>>>(Wk, Wq, Wv, Wo, wkb);
  attn_fused_kernel<<<BB * HH * WW, 256, LDS_BYTES, stream>>>(
      x, sA, sB, wkb, wqb, wvb, wob, out);
}
